// CrossAttention_17162689315176
// MI455X (gfx1250) — matrix operand order repaired and checked by reading, not compiled, not run on hardware
//
#include <hip/hip_runtime.h>
#include <hip/hip_bf16.h>

// ---------------------------------------------------------------------------
// CDNA5 (gfx1250) windowed cross-attention, bf16 WMMA pipeline.
// 1024 windows x (16 queries, 64 keys), 8 heads x d=64, C=512.
// All matrix math via v_wmma_f32_16x16x32_bf16 (wave32).
// GEMMs use 4x2 register blocking (64x32 out-tile per wave): 8 wmma per
// 6 fragment loads per 32-K step, so the loop is WMMA-bound, and the packed
// bf16 operands (<= 67 MB) live in the 192 MB L2.
// ---------------------------------------------------------------------------

typedef __attribute__((ext_vector_type(16))) __bf16 v16bf;
typedef __attribute__((ext_vector_type(8)))  float  v8f;

struct B32x2 { uint4 lo, hi; };   // 32 bytes == one v16bf fragment

static __device__ inline unsigned short f2bf_raw(float f) {
  unsigned u = __builtin_bit_cast(unsigned, f);
  u += 0x7FFFu + ((u >> 16) & 1u);          // round-to-nearest-even
  return (unsigned short)(u >> 16);
}
static __device__ inline __bf16 f2bf(float f) {
  unsigned short s = f2bf_raw(f);
  return __builtin_bit_cast(__bf16, s);
}

// A-fragment (16x32 bf16, M x K). contiguous K order (same as B here): lane L holds row m=L&15;
// lane group g=L>>4: elements j=0..15 -> K = g*16+j.
// p points at element (row0, k0) of a row-major [M][ld] matrix.
static __device__ inline v16bf load_a_frag(const __bf16* p, int ld, int lane) {
  int m = lane & 15, g = lane >> 4;
  const __bf16* q = p + (size_t)m * ld + g * 16;
  B32x2 t;
  t.lo = *(const uint4*)(q);
  t.hi = *(const uint4*)(q + 8);
  return __builtin_bit_cast(v16bf, t);
}

// B-fragment (32x16 bf16, K x N) supplied as B^T row-major [N][ld] (K contiguous).
// Lane L holds column n=L&15; elements j: K = (L>>4)*16 + j -> one 32B load.
static __device__ inline v16bf load_b_frag(const __bf16* p, int ld, int lane) {
  int n = lane & 15, g = lane >> 4;
  const __bf16* q = p + (size_t)n * ld + g * 16;
  return *(const v16bf*)q;
}

static __device__ inline v8f wmma_bf16(v16bf a, v16bf b, v8f c) {
  return __builtin_amdgcn_wmma_f32_16x16x32_bf16(false, a, false, b, (short)0, c,
                                                 false, false);
}

// 64x32 output tile per wave: 4 M-subtiles x 2 N-subtiles.
// A row-major [..][K] at the 64-row group base, BT row-major [..][K] at the
// 32-col group base. acc[mi][ni] are the 8 16x16 accumulators.
static __device__ inline void gemm_tile_4x2(const __bf16* Ab, const __bf16* Bb,
                                            int K, int lane, v8f acc[4][2]) {
  for (int k = 0; k < K; k += 32) {
    if (k + 32 < K)
      __builtin_prefetch(Ab + k + 32, 0, 1);       // global_prefetch_b8
    v16bf b0 = load_b_frag(Bb + k, K, lane);
    v16bf b1 = load_b_frag(Bb + (size_t)16 * K + k, K, lane);
#pragma unroll
    for (int mi = 0; mi < 4; ++mi) {
      v16bf a = load_a_frag(Ab + (size_t)mi * 16 * K + k, K, lane);
      acc[mi][0] = wmma_bf16(a, b0, acc[mi][0]);
      acc[mi][1] = wmma_bf16(a, b1, acc[mi][1]);
    }
  }
}

// ---------------------------------------------------------------------------
// Packing kernels
// ---------------------------------------------------------------------------

// W[in][out] f32 -> WT[out][in] bf16 (512x512)
__global__ void k_pack_wT(const float* __restrict__ W, __bf16* __restrict__ WT) {
  int i = blockIdx.x * 256 + threadIdx.x;      // 262144
  int o = i >> 9, in = i & 511;
  WT[i] = f2bf(W[in * 512 + o]);
}

// x (4,512,4,32,32) f32 -> Xq[(w*16+t)][c] bf16, window order
__global__ void k_pack_q(const float* __restrict__ x, __bf16* __restrict__ Xq) {
  size_t i = (size_t)blockIdx.x * 256 + threadIdx.x;  // 8388608
  int ww = (int)(i & 31), hh = (int)((i >> 5) & 31);
  int tt = (int)((i >> 10) & 3), c = (int)((i >> 12) & 511), b = (int)(i >> 21);
  int w = (b * 16 + (hh >> 1)) * 16 + (ww >> 1);
  int t = tt * 4 + (hh & 1) * 2 + (ww & 1);
  Xq[(size_t)(w * 16 + t) * 512 + c] = f2bf(x[i]);
}

// context (4,512,128,128) f32 -> Ctx[(w*64+t)][c] bf16, window order
__global__ void k_pack_ctx(const float* __restrict__ ctx, __bf16* __restrict__ C) {
  size_t i = (size_t)blockIdx.x * 256 + threadIdx.x;  // 33554432
  int xx = (int)(i & 127), yy = (int)((i >> 7) & 127);
  int c = (int)((i >> 14) & 511), b = (int)(i >> 23);
  int w = (b * 16 + (yy >> 3)) * 16 + (xx >> 3);
  int t = (yy & 7) * 8 + (xx & 7);
  C[(size_t)(w * 64 + t) * 512 + c] = f2bf(ctx[i]);
}

// ---------------------------------------------------------------------------
// GEMM kernels: one 64x32 out-tile per wave, 8 waves/block
// ---------------------------------------------------------------------------

__global__ void k_gemm_bf16(const __bf16* __restrict__ A,
                            const __bf16* __restrict__ BT,
                            __bf16* __restrict__ D, int N, int K) {
  int wave = (blockIdx.x * blockDim.x + threadIdx.x) >> 5;
  int lane = threadIdx.x & 31;
  int ng = N >> 5;                               // 32-col groups
  int tm = wave / ng, tn = wave - tm * ng;
  v8f acc[4][2] = {};
  gemm_tile_4x2(A + (size_t)tm * 64 * K, BT + (size_t)tn * 32 * K, K, lane, acc);
  int n = lane & 15, g = lane >> 4;
#pragma unroll
  for (int mi = 0; mi < 4; ++mi)
#pragma unroll
    for (int ni = 0; ni < 2; ++ni)
#pragma unroll
      for (int r = 0; r < 8; ++r)
        D[(size_t)(tm * 64 + mi * 16 + r + 8 * g) * N + tn * 32 + ni * 16 + n] =
            f2bf(acc[mi][ni][r]);
}

// Same, but stores D transposed: DT[N][M] (used for V^T). M = token count.
__global__ void k_gemm_bf16_T(const __bf16* __restrict__ A,
                              const __bf16* __restrict__ BT,
                              __bf16* __restrict__ DT, int M, int N, int K) {
  int wave = (blockIdx.x * blockDim.x + threadIdx.x) >> 5;
  int lane = threadIdx.x & 31;
  int ng = N >> 5;
  int tm = wave / ng, tn = wave - tm * ng;
  v8f acc[4][2] = {};
  gemm_tile_4x2(A + (size_t)tm * 64 * K, BT + (size_t)tn * 32 * K, K, lane, acc);
  int n = lane & 15, g = lane >> 4;
#pragma unroll
  for (int mi = 0; mi < 4; ++mi)
#pragma unroll
    for (int ni = 0; ni < 2; ++ni) {
      unsigned short h[8];
#pragma unroll
      for (int r = 0; r < 8; ++r) h[r] = f2bf_raw(acc[mi][ni][r]);
      uint4 pk = make_uint4((unsigned)h[0] | ((unsigned)h[1] << 16),
                            (unsigned)h[2] | ((unsigned)h[3] << 16),
                            (unsigned)h[4] | ((unsigned)h[5] << 16),
                            (unsigned)h[6] | ((unsigned)h[7] << 16));
      // element (m=r+8g, n): DT[ch][tok], tok contiguous over r
      *(uint4*)(DT + (size_t)(tn * 32 + ni * 16 + n) * M + tm * 64 + mi * 16 +
                8 * g) = pk;
    }
}

// ---------------------------------------------------------------------------
// Fused per-window attention: 1024 blocks, 8 waves/block, wave == head.
// Qb[16384][512], Kb[65536][512] token-major; Vt[512][65536] channel-major.
// ---------------------------------------------------------------------------

__global__ void k_attention(const __bf16* __restrict__ Qb,
                            const __bf16* __restrict__ Kb,
                            const __bf16* __restrict__ Vt,
                            __bf16* __restrict__ Ob) {
  __shared__ float Ssm[8][16][64];                  // 32 KB
  __shared__ __align__(32) __bf16 Psm[8][16][64];   // 16 KB
  int w = blockIdx.x;
  int h = threadIdx.x >> 5;      // head == wave id
  int lane = threadIdx.x & 31;
  int n = lane & 15, g = lane >> 4;

  // Q A-fragments (16x64 split into two 16x32): rows = queries, K = channels
  const __bf16* Qp = Qb + (size_t)(w * 16) * 512 + h * 64;
  v16bf q0 = load_a_frag(Qp, 512, lane);
  v16bf q1 = load_a_frag(Qp + 32, 512, lane);

  // S = (Q K^T) * d^-1/2 ; K token-major is exactly B^T for this GEMM
#pragma unroll
  for (int n0 = 0; n0 < 64; n0 += 16) {
    const __bf16* Kp = Kb + (size_t)(w * 64 + n0) * 512 + h * 64;
    v16bf b0 = load_b_frag(Kp, 512, lane);
    v16bf b1 = load_b_frag(Kp + 32, 512, lane);
    v8f s = {};
    s = wmma_bf16(q0, b0, s);
    s = wmma_bf16(q1, b1, s);
#pragma unroll
    for (int r = 0; r < 8; ++r)
      Ssm[h][r + 8 * g][n0 + n] = s[r] * 0.125f;    // 1/sqrt(64)
  }

  // Softmax over 64 keys. Wave-private LDS region; DS ops are in-order per
  // wave so no barrier is needed. Lane pair (2 lanes) per row.
  int m = lane >> 1, half = lane & 1;
  float mx = -3.4e38f;
  for (int k = 0; k < 32; ++k) mx = fmaxf(mx, Ssm[h][m][half * 32 + k]);
  mx = fmaxf(mx, __shfl_xor(mx, 1));
  float sum = 0.f;
  for (int k = 0; k < 32; ++k) sum += __expf(Ssm[h][m][half * 32 + k] - mx);
  sum += __shfl_xor(sum, 1);
  float inv = 1.0f / sum;
  for (int k = 0; k < 32; ++k)
    Psm[h][m][half * 32 + k] = f2bf(__expf(Ssm[h][m][half * 32 + k] - mx) * inv);

  // Reload P in A-fragment layout (LDS transpose C-layout -> A-layout)
  const __bf16* Pp = &Psm[h][0][0];
  v16bf p0 = load_a_frag(Pp, 64, lane);
  v16bf p1 = load_a_frag(Pp + 32, 64, lane);

  // O = P V ; Vt[ch][tok] is B^T (K = key tokens contiguous)
#pragma unroll
  for (int ct = 0; ct < 4; ++ct) {
    const __bf16* Vp = Vt + (size_t)(h * 64 + ct * 16) * 65536 + (size_t)w * 64;
    v16bf vb0 = load_b_frag(Vp, 65536, lane);
    v16bf vb1 = load_b_frag(Vp + 32, 65536, lane);
    v8f o = {};
    o = wmma_bf16(p0, vb0, o);
    o = wmma_bf16(p1, vb1, o);
#pragma unroll
    for (int r = 0; r < 8; ++r)
      Ob[(size_t)(w * 16 + r + 8 * g) * 512 + h * 64 + ct * 16 + n] = f2bf(o[r]);
  }
}

// ---------------------------------------------------------------------------
// Output projection + bias, scatter to (B, T*H*W, C) f32
// ---------------------------------------------------------------------------

__global__ void k_proj_out(const __bf16* __restrict__ Ob,
                           const __bf16* __restrict__ WoT,
                           const float* __restrict__ bias,
                           float* __restrict__ out) {
  int wave = (blockIdx.x * blockDim.x + threadIdx.x) >> 5;
  int lane = threadIdx.x & 31;
  int tm = wave >> 4, tn = wave & 15;            // N/32 == 16 col groups
  v8f acc[4][2] = {};
  gemm_tile_4x2(Ob + (size_t)tm * 64 * 512, WoT + (size_t)tn * 32 * 512, 512,
                lane, acc);
  int n = lane & 15, g = lane >> 4;
#pragma unroll
  for (int ni = 0; ni < 2; ++ni) {
    int col = tn * 32 + ni * 16 + n;
    float bv = bias[col];
#pragma unroll
    for (int mi = 0; mi < 4; ++mi)
#pragma unroll
      for (int r = 0; r < 8; ++r) {
        int grow = tm * 64 + mi * 16 + r + 8 * g;  // window-order token index
        int w = grow >> 4, t = grow & 15;
        int b = w >> 8, hy = (w >> 4) & 15, wx = w & 15;
        int tt = t >> 2, iy = (t >> 1) & 1, ix = t & 1;
        int row = tt * 1024 + (hy * 2 + iy) * 32 + (wx * 2 + ix);
        out[((size_t)b * 4096 + row) * 512 + col] = acc[mi][ni][r] + bv;
      }
  }
}

// ---------------------------------------------------------------------------

extern "C" void kernel_launch(void* const* d_in, const int* in_sizes, int n_in,
                              void* d_out, int out_size, void* d_ws, size_t ws_size,
                              hipStream_t stream) {
  const float* x   = (const float*)d_in[0];   // (4,512,4,32,32)
  const float* ctx = (const float*)d_in[1];   // (4,512,128,128)
  const float* Wq  = (const float*)d_in[2];
  const float* Wk  = (const float*)d_in[3];
  const float* Wv  = (const float*)d_in[4];
  const float* Wo  = (const float*)d_in[5];
  const float* bo  = (const float*)d_in[6];
  float* out = (float*)d_out;                  // (4, 4096, 512)

  char* ws = (char*)d_ws;
  size_t off = 0;
  auto carve = [&](size_t bytes) {
    char* p = ws + off;
    off = (off + bytes + 255) & ~(size_t)255;
    return p;
  };
  __bf16* WqT = (__bf16*)carve(512 * 512 * 2);
  __bf16* WkT = (__bf16*)carve(512 * 512 * 2);
  __bf16* WvT = (__bf16*)carve(512 * 512 * 2);
  __bf16* WoT = (__bf16*)carve(512 * 512 * 2);
  __bf16* Xq  = (__bf16*)carve((size_t)16384 * 512 * 2);  // packed q tokens
  __bf16* Ctx = (__bf16*)carve((size_t)65536 * 512 * 2);  // packed ctx tokens
  __bf16* Qb  = (__bf16*)carve((size_t)16384 * 512 * 2);  // Q   [tok][ch]
  __bf16* Kb  = (__bf16*)carve((size_t)65536 * 512 * 2);  // K   [tok][ch]
  __bf16* Vt  = (__bf16*)carve((size_t)65536 * 512 * 2);  // V^T [ch][tok]
  __bf16* Ob  = (__bf16*)carve((size_t)16384 * 512 * 2);  // attn out [tok][ch]

  // 1) weights -> bf16 transposed
  k_pack_wT<<<1024, 256, 0, stream>>>(Wq, WqT);
  k_pack_wT<<<1024, 256, 0, stream>>>(Wk, WkT);
  k_pack_wT<<<1024, 256, 0, stream>>>(Wv, WvT);
  k_pack_wT<<<1024, 256, 0, stream>>>(Wo, WoT);

  // 2) activations -> token-major bf16 (window order)
  k_pack_q  <<<32768,  256, 0, stream>>>(x, Xq);
  k_pack_ctx<<<131072, 256, 0, stream>>>(ctx, Ctx);

  // 3) projections (64x32 tile per wave, 8 waves per block)
  // Q = Xq @ Wq : (16384/64)*(512/32) = 4096 waves -> 512 blocks
  k_gemm_bf16<<<512, 256, 0, stream>>>(Xq, WqT, Qb, 512, 512);
  // K = Ctx @ Wk : (65536/64)*16 = 16384 waves -> 2048 blocks
  k_gemm_bf16<<<2048, 256, 0, stream>>>(Ctx, WkT, Kb, 512, 512);
  // V^T = (Ctx @ Wv)^T stored channel-major
  k_gemm_bf16_T<<<2048, 256, 0, stream>>>(Ctx, WvT, Vt, 65536, 512, 512);

  // 4) fused attention per window (wave == head)
  k_attention<<<1024, 256, 0, stream>>>(Qb, Kb, Vt, Ob);

  // 5) output projection + bias + scatter to final layout
  k_proj_out<<<512, 256, 0, stream>>>(Ob, WoT, bo, out);
}
